// ResMLPMoEBlock_49924699848919
// MI455X (gfx1250) — matrix operand order repaired and checked by reading, not compiled, not run on hardware
//
#include <hip/hip_runtime.h>

#define NTOK 8192
#define DIM  512
#define MHID 2048
#define EHID 1024
#define NEXP 8

typedef __attribute__((ext_vector_type(16))) __bf16 v16bf;
typedef __attribute__((ext_vector_type(8)))  float  v8f;
typedef __attribute__((ext_vector_type(4)))  unsigned int v4u;
typedef __attribute__((ext_vector_type(8)))  unsigned int v8u;

// ---------- helpers ----------

static __device__ __forceinline__ __bf16 f2bf(float f) {
  unsigned u = __builtin_bit_cast(unsigned, f);
  u += 0x7FFFu + ((u >> 16) & 1u);            // round-to-nearest-even
  unsigned short h = (unsigned short)(u >> 16);
  return __builtin_bit_cast(__bf16, h);
}

static __device__ __forceinline__ float gelu_exact(float x) {
  return 0.5f * x * (1.0f + erff(x * 0.7071067811865476f));
}

static __device__ __forceinline__ v8f wmma_bf16(v16bf a, v16bf b, v8f c) {
  return __builtin_amdgcn_wmma_f32_16x16x32_bf16(
      /*neg_a=*/false, a, /*neg_b=*/false, b,
      /*c_mod=*/(short)0, c, /*reuse_a=*/false, /*reuse_b=*/false);
}

// CDNA5 async copy: 16B global -> LDS, tracked by ASYNCcnt, no VGPR data path.
static __device__ __forceinline__ void async_b128(__bf16* lds_dst, const __bf16* gsrc) {
  unsigned ldsa = (unsigned)(uintptr_t)lds_dst;
  asm volatile("global_load_async_to_lds_b128 %0, %1, off"
               :: "v"(ldsa), "v"(gsrc)
               : "memory");
}
static __device__ __forceinline__ void wait_async0() {
  asm volatile("s_wait_asynccnt 0" ::: "memory");
}

// CDNA5 Tensor Data Mover: one DMA moves a 2D tile (tile_d1 rows x tile_d0
// bf16 elements, row pitch d0_stride elements) from global to LDS, packed
// row-major at lds_addr.  D# per ISA 8.3/8.4, 2-group form (<=2D tensor,
// VADDR2/3 = NULL).  Descriptor is wave-uniform -> lives in SGPR tuples.
static __device__ __forceinline__ void tdm_load_2d(unsigned lds_addr,
                                                   const void* gptr,
                                                   unsigned tensor_d0,
                                                   unsigned tensor_d1,
                                                   unsigned tile_d0,
                                                   unsigned tile_d1,
                                                   unsigned d0_stride) {
  unsigned long long ga = (unsigned long long)(uintptr_t)gptr;
  v4u g0;
  g0[0] = 1u;                                   // count=1 (valid), user mode
  g0[1] = lds_addr;                             // LDS byte address
  g0[2] = (unsigned)ga;                         // global_addr[31:0]
  g0[3] = (unsigned)(ga >> 32) | (2u << 30);    // global_addr[56:32] | type=2
  v8u g1;
  g1[0] = (1u << 16);                           // wg_mask=0 | data_size=1 (2B)
  g1[1] = (tensor_d0 & 0xFFFFu) << 16;          // atomic_barrier=0 | tdim0.lo
  g1[2] = (tensor_d0 >> 16) | ((tensor_d1 & 0xFFFFu) << 16);  // tdim0.hi|tdim1.lo
  g1[3] = (tensor_d1 >> 16) | (tile_d0 << 16);  // tdim1.hi | tile_dim0
  g1[4] = tile_d1 & 0xFFFFu;                    // tile_dim1 | tile_dim2=0
  g1[5] = d0_stride;                            // tensor_dim0_stride[31:0]
  g1[6] = 0u;                                   // stride[47:32] | dim1_stride.lo
  g1[7] = 0u;                                   // dim1_stride.hi (unused, 2D)
  asm volatile("tensor_load_to_lds %0, %1" :: "s"(g0), "s"(g1) : "memory");
}
static __device__ __forceinline__ void wait_tensor0() {
  __builtin_amdgcn_s_wait_tensorcnt(0);
}

// A fragment: 16x32 bf16 tile, row-major in LDS with stride lda.
static __device__ __forceinline__ v16bf frag_a(const __bf16* t, int lane, int lda) {
  const __bf16* p = t + (lane & 15) * lda;
  const int kb = (lane >> 4) << 3;
  v16bf a;
#pragma unroll
  for (int i = 0; i < 8; ++i) a[i] = p[kb + i];
#pragma unroll
  for (int i = 0; i < 8; ++i) a[8 + i] = p[16 + kb + i];
  return a;
}

// B fragment from LDS tile stored TRANSPOSED as Bt[n][k] (16 n-rows, k=32).
static __device__ __forceinline__ v16bf frag_bt(const __bf16* t, int lane) {
  const __bf16* p = t + (lane & 15) * 32 + ((lane >> 4) << 3);
  v16bf b;
#pragma unroll
  for (int i = 0; i < 8; ++i) b[i] = p[i];
#pragma unroll
  for (int i = 0; i < 8; ++i) b[8 + i] = p[16 + i];
  return b;
}

// block-wide row mean / rsqrt(var+eps); red must be __shared__ float[18]
static __device__ __forceinline__ void row_stats(const float* xr, int n, float* red,
                                                 float& mean, float& inv) {
  const int tid = threadIdx.x, lane = tid & 31, wave = tid >> 5;
  float s = 0.f, ss = 0.f;
  for (int i = tid; i < n; i += 256) { float v = xr[i]; s += v; ss += v * v; }
#pragma unroll
  for (int off = 16; off > 0; off >>= 1) {
    s  += __shfl_xor(s,  off, 32);
    ss += __shfl_xor(ss, off, 32);
  }
  if (lane == 0) { red[wave] = s; red[8 + wave] = ss; }
  __syncthreads();
  if (tid == 0) {
    float S = 0.f, SS = 0.f;
    for (int w = 0; w < 8; ++w) { S += red[w]; SS += red[8 + w]; }
    float m = S / n;
    red[16] = m;
    red[17] = SS / n - m * m;
  }
  __syncthreads();
  mean = red[16];
  inv  = rsqrtf(red[17] + 1e-5f);
}

// ---------- small kernels ----------

// transpose+convert: in [batch][rows][cols] f32 -> out [batch][cols][rows] bf16
__global__ __launch_bounds__(256) void cvt_tr_bf16_k(const float* __restrict__ in,
                                                     __bf16* __restrict__ out,
                                                     int rows, int cols, int batch) {
  const long total = (long)batch * rows * cols;
  for (long i = (long)blockIdx.x * 256 + threadIdx.x; i < total;
       i += (long)gridDim.x * 256) {
    long b = i / ((long)rows * cols);
    long r = (i / cols) % rows;
    long c = i % cols;
    out[b * (long)rows * cols + c * rows + r] = f2bf(in[i]);
  }
}

__global__ __launch_bounds__(256) void ln1_k(const float* __restrict__ x,
                                             const float* __restrict__ g,
                                             const float* __restrict__ b,
                                             __bf16* __restrict__ y) {
  __shared__ float red[18];
  const int row = blockIdx.x, tid = threadIdx.x;
  const float* xr = x + (size_t)row * DIM;
  float m, inv;
  row_stats(xr, DIM, red, m, inv);
  for (int i = tid; i < DIM; i += 256)
    y[(size_t)row * DIM + i] = f2bf((xr[i] - m) * inv * g[i] + b[i]);
}

// x3 = LN(x2)*g + b + x2 ; writes fp32 and bf16 copies
__global__ __launch_bounds__(256) void ln2_k(const float* __restrict__ x2,
                                             const float* __restrict__ g,
                                             const float* __restrict__ b,
                                             float* __restrict__ x3,
                                             __bf16* __restrict__ x3b) {
  __shared__ float red[18];
  const int row = blockIdx.x, tid = threadIdx.x;
  const float* xr = x2 + (size_t)row * DIM;
  float m, inv;
  row_stats(xr, DIM, red, m, inv);
  for (int i = tid; i < DIM; i += 256) {
    float v = (xr[i] - m) * inv * g[i] + b[i] + xr[i];
    x3[(size_t)row * DIM + i]  = v;
    x3b[(size_t)row * DIM + i] = f2bf(v);
  }
}

__global__ __launch_bounds__(256) void lnf_k(const float* __restrict__ a,
                                             const float* __restrict__ g,
                                             const float* __restrict__ b,
                                             float* __restrict__ out) {
  __shared__ float red[18];
  const int row = blockIdx.x, tid = threadIdx.x;
  const float* xr = a + (size_t)row * DIM;
  float m, inv;
  row_stats(xr, DIM, red, m, inv);
  for (int i = tid; i < DIM; i += 256)
    out[(size_t)row * DIM + i] = (xr[i] - m) * inv * g[i] + b[i];
  if (row == 0 && tid == 0) out[(size_t)NTOK * DIM] = 0.f;  // aux_loss
}

// per-token router: softmax over 8 logits, top-2 renormalized dense gates
__global__ __launch_bounds__(256) void router_k(const float* __restrict__ x3,
                                                const float* __restrict__ wr,
                                                float* __restrict__ gates) {
  const int n = blockIdx.x * 256 + threadIdx.x;
  if (n >= NTOK) return;
  const float* xr = x3 + (size_t)n * DIM;
  float lg[NEXP];
#pragma unroll
  for (int e = 0; e < NEXP; ++e) {
    const float* w = wr + (size_t)e * DIM;
    float s = 0.f;
    for (int d = 0; d < DIM; ++d) s += xr[d] * w[d];
    lg[e] = s;
  }
  float mx = lg[0];
#pragma unroll
  for (int e = 1; e < NEXP; ++e) mx = fmaxf(mx, lg[e]);
  float den = 0.f;
#pragma unroll
  for (int e = 0; e < NEXP; ++e) { lg[e] = __expf(lg[e] - mx); den += lg[e]; }
#pragma unroll
  for (int e = 0; e < NEXP; ++e) lg[e] /= den;
  int i1 = 0;
#pragma unroll
  for (int e = 1; e < NEXP; ++e) if (lg[e] > lg[i1]) i1 = e;
  int i2 = (i1 == 0) ? 1 : 0;
#pragma unroll
  for (int e = 0; e < NEXP; ++e) if (e != i1 && lg[e] > lg[i2]) i2 = e;
  float sum = lg[i1] + lg[i2];
#pragma unroll
  for (int e = 0; e < NEXP; ++e) gates[(size_t)n * NEXP + e] = 0.f;
  gates[(size_t)n * NEXP + i1] = lg[i1] / sum;
  gates[(size_t)n * NEXP + i2] = lg[i2] / sum;
}

// ---------- WMMA GEMM: C[M,Nn] = A[M,K] @ BT[Nn,K]^T + bias ----------
// Staging via async-to-LDS with ping-pong double buffering so loads overlap
// v_wmma issue.  mode 0: outb = bf16(gelu(C))   mode 1: outf = C + resid
__global__ __launch_bounds__(256) void gemm_bf16(
    const __bf16* __restrict__ A, const __bf16* __restrict__ BT,
    const float* __restrict__ bias, const float* __restrict__ resid,
    __bf16* __restrict__ outb, float* __restrict__ outf,
    int M, int K, int Nn, int mode) {
  __shared__ __align__(16) __bf16 As[2][128 * 32];
  __shared__ __align__(16) __bf16 Bt[2][128 * 32];
  const int tid = threadIdx.x, lane = tid & 31, wave = tid >> 5;
  const int m0 = blockIdx.y * 128, n0 = blockIdx.x * 128;

  v8f acc[8];
#pragma unroll
  for (int i = 0; i < 8; ++i) acc[i] = (v8f){0.f,0.f,0.f,0.f,0.f,0.f,0.f,0.f};

  const int srow = tid >> 1, skh = (tid & 1) << 4;   // 128 rows x 32, 32B/thread

  auto stage = [&](int buf, int k0) {
    const __bf16* ag = A + (size_t)(m0 + srow) * K + k0 + skh;
    __bf16* as = &As[buf][srow * 32 + skh];
    async_b128(as,     ag);
    async_b128(as + 8, ag + 8);
    const __bf16* bg = BT + (size_t)(n0 + srow) * K + k0 + skh;
    __bf16* bs = &Bt[buf][srow * 32 + skh];
    async_b128(bs,     bg);
    async_b128(bs + 8, bg + 8);
    if (k0 + 32 < K) __builtin_prefetch(ag + 32, 0, 0);   // global_prefetch_b8
  };

  stage(0, 0);
  wait_async0();
  __syncthreads();

  int cur = 0;
  for (int k0 = 0; k0 < K; k0 += 32) {
    if (k0 + 32 < K) stage(cur ^ 1, k0 + 32);   // async prefetch next slab
    v16bf a = frag_a(&As[cur][wave * 16 * 32], lane, 32);
#pragma unroll
    for (int cs = 0; cs < 8; ++cs) {
      v16bf b = frag_bt(&Bt[cur][cs * 16 * 32], lane);
      acc[cs] = wmma_bf16(a, b, acc[cs]);
    }
    wait_async0();
    __syncthreads();
    cur ^= 1;
  }

  const int nlo = lane & 15, mb = (lane >> 4) << 3;
  const int rowbase = m0 + wave * 16 + mb;
#pragma unroll
  for (int cs = 0; cs < 8; ++cs) {
    const int col = n0 + cs * 16 + nlo;
    const float bv = bias[col];
#pragma unroll
    for (int r = 0; r < 8; ++r) {
      const size_t idx = (size_t)(rowbase + r) * Nn + col;
      float v = acc[cs][r] + bv;
      if (mode == 0) outb[idx] = f2bf(gelu_exact(v));
      else           outf[idx] = v + resid[idx];
    }
  }
}

// ---------- fused MoE expert: per (64-token tile, expert) ----------
// All tile staging done by the Tensor Data Mover: wave 0 issues one
// tensor_load_to_lds per tile (TENSORcnt), waits, and the workgroup barrier
// publishes LDS.  he = gelu(x3b @ ew1[e] + eb1[e]) lives in LDS (EH chunks of
// 256);  out += gate[n,e] * (he @ ew2[e] + eb2[e]) via atomicAdd.
__global__ __launch_bounds__(256) void expert_moe(
    const __bf16* __restrict__ x3b, const __bf16* __restrict__ ew1bT,
    const float* __restrict__ eb1, const __bf16* __restrict__ ew2bT,
    const float* __restrict__ eb2, const float* __restrict__ gates,
    float* __restrict__ outacc) {
  __shared__ __align__(16) __bf16 he[64 * 256];   // 32 KB
  __shared__ __align__(16) __bf16 As[64 * 32];    // 4 KB
  __shared__ __align__(16) __bf16 Bt[128 * 32];   // 8 KB
  const int e  = blockIdx.y;
  const int m0 = blockIdx.x * 64;
  const int tid = threadIdx.x, lane = tid & 31, wave = tid >> 5;
  const int rs = wave & 3;            // row strip (16 rows)
  const int cw = (wave >> 2) << 2;    // col strip base (4 subtiles)
  const __bf16* W1T = ew1bT + (size_t)e * EHID * DIM;   // [1024][512]
  const __bf16* W2T = ew2bT + (size_t)e * DIM * EHID;   // [512][1024]

  const unsigned lds_as = (unsigned)(uintptr_t)As;
  const unsigned lds_bt = (unsigned)(uintptr_t)Bt;

  v8f acc2[4][4];
#pragma unroll
  for (int d = 0; d < 4; ++d)
#pragma unroll
    for (int c = 0; c < 4; ++c) acc2[d][c] = (v8f){0.f,0.f,0.f,0.f,0.f,0.f,0.f,0.f};

  const int nlo = lane & 15, mb = (lane >> 4) << 3;

  for (int q = 0; q < 4; ++q) {          // EH chunk [q*256, q*256+256)
    const int eh0 = q * 256;
    // ---- stage 1: he[64][256] = gelu(x3b_tile @ W1[:,chunk] + eb1)
    for (int cb = 0; cb < 2; ++cb) {
      v8f acc1[4];
#pragma unroll
      for (int c = 0; c < 4; ++c) acc1[c] = (v8f){0.f,0.f,0.f,0.f,0.f,0.f,0.f,0.f};
      for (int k0 = 0; k0 < DIM; k0 += 32) {
        if (wave == 0) {
          // A tile: 64 rows x 32 of x3b; B tile: 128 rows x 32 of W1T
          tdm_load_2d(lds_as, x3b + (size_t)m0 * DIM + k0,
                      DIM, 64, 32, 64, DIM);
          tdm_load_2d(lds_bt, W1T + (size_t)(eh0 + cb * 128) * DIM + k0,
                      DIM, 128, 32, 128, DIM);
          wait_tensor0();
        }
        __syncthreads();
        v16bf a = frag_a(&As[rs * 16 * 32], lane, 32);
#pragma unroll
        for (int c = 0; c < 4; ++c) {
          v16bf b = frag_bt(&Bt[(cw + c) * 16 * 32], lane);
          acc1[c] = wmma_bf16(a, b, acc1[c]);
        }
        __syncthreads();
      }
#pragma unroll
      for (int c = 0; c < 4; ++c) {
        const int col = cb * 128 + (cw + c) * 16 + nlo;
        const float bv = eb1[(size_t)e * EHID + eh0 + col];
#pragma unroll
        for (int r = 0; r < 8; ++r)
          he[(rs * 16 + mb + r) * 256 + col] = f2bf(gelu_exact(acc1[c][r] + bv));
      }
      __syncthreads();
    }
    // ---- stage 2: acc2 += he @ W2[chunk, :]
    for (int db = 0; db < 4; ++db) {
      for (int k0 = 0; k0 < 256; k0 += 32) {
        if (wave == 0) {
          // B tile: 128 rows x 32 of W2T (row pitch EHID)
          tdm_load_2d(lds_bt, W2T + (size_t)(db * 128) * EHID + eh0 + k0,
                      EHID, 128, 32, 128, EHID);
          wait_tensor0();
        }
        __syncthreads();
        v16bf a = frag_a(he + rs * 16 * 256 + k0, lane, 256);
#pragma unroll
        for (int c = 0; c < 4; ++c) {
          v16bf b = frag_bt(&Bt[(cw + c) * 16 * 32], lane);
          acc2[db][c] = wmma_bf16(a, b, acc2[db][c]);
        }
        __syncthreads();
      }
    }
    __syncthreads();   // he reads done before next chunk overwrites it
  }

#pragma unroll
  for (int db = 0; db < 4; ++db) {
#pragma unroll
    for (int c = 0; c < 4; ++c) {
      const int col = db * 128 + (cw + c) * 16 + nlo;
      const float bv = eb2[(size_t)e * DIM + col];
#pragma unroll
      for (int r = 0; r < 8; ++r) {
        const int grow = m0 + rs * 16 + mb + r;
        const float g = gates[(size_t)grow * NEXP + e];
        atomicAdd(outacc + (size_t)grow * DIM + col, (acc2[db][c][r] + bv) * g);
      }
    }
  }
}

// ---------- launch ----------

extern "C" void kernel_launch(void* const* d_in, const int* in_sizes, int n_in,
                              void* d_out, int out_size, void* d_ws, size_t ws_size,
                              hipStream_t stream) {
  (void)in_sizes; (void)n_in; (void)out_size; (void)ws_size;
  const float* x    = (const float*)d_in[0];
  const float* ln1g = (const float*)d_in[1];
  const float* ln1b = (const float*)d_in[2];
  const float* w1   = (const float*)d_in[3];
  const float* b1   = (const float*)d_in[4];
  const float* w2   = (const float*)d_in[5];
  const float* b2   = (const float*)d_in[6];
  const float* ln2g = (const float*)d_in[7];
  const float* ln2b = (const float*)d_in[8];
  const float* wr   = (const float*)d_in[9];
  const float* ew1  = (const float*)d_in[10];
  const float* eb1  = (const float*)d_in[11];
  const float* ew2  = (const float*)d_in[12];
  const float* eb2  = (const float*)d_in[13];
  const float* lnfg = (const float*)d_in[14];
  const float* lnfb = (const float*)d_in[15];
  float* out = (float*)d_out;

  char* ws = (char*)d_ws;
  size_t off = 0;
  auto alloc = [&](size_t bytes) {
    char* p = ws + off;
    off += (bytes + 255) & ~(size_t)255;
    return p;
  };
  __bf16* xb     = (__bf16*)alloc((size_t)NTOK * DIM * 2);
  __bf16* w1bT   = (__bf16*)alloc((size_t)MHID * DIM * 2);        // [MH][D]
  __bf16* w2bT   = (__bf16*)alloc((size_t)DIM * MHID * 2);        // [D][MH]
  __bf16* ew1bT  = (__bf16*)alloc((size_t)NEXP * EHID * DIM * 2); // [E][EH][D]
  __bf16* ew2bT  = (__bf16*)alloc((size_t)NEXP * DIM * EHID * 2); // [E][D][EH]
  __bf16* h1     = (__bf16*)alloc((size_t)NTOK * MHID * 2);
  float*  x2     = (float*)alloc((size_t)NTOK * DIM * 4);
  float*  x3     = (float*)alloc((size_t)NTOK * DIM * 4);
  __bf16* x3b    = (__bf16*)alloc((size_t)NTOK * DIM * 2);
  float*  gates  = (float*)alloc((size_t)NTOK * NEXP * 4);
  float*  outacc = (float*)alloc((size_t)NTOK * DIM * 4);

  const int cw1 = DIM * MHID;          // 1M elems per matrix
  const int cew = NEXP * DIM * EHID;   // 4M elems per tensor
  cvt_tr_bf16_k<<<(cw1 + 255) / 256, 256, 0, stream>>>(w1, w1bT, DIM, MHID, 1);
  cvt_tr_bf16_k<<<(cw1 + 255) / 256, 256, 0, stream>>>(w2, w2bT, MHID, DIM, 1);
  cvt_tr_bf16_k<<<(cew + 255) / 256, 256, 0, stream>>>(ew1, ew1bT, DIM, EHID, NEXP);
  cvt_tr_bf16_k<<<(cew + 255) / 256, 256, 0, stream>>>(ew2, ew2bT, EHID, DIM, NEXP);

  ln1_k<<<NTOK, 256, 0, stream>>>(x, ln1g, ln1b, xb);

  gemm_bf16<<<dim3(MHID / 128, NTOK / 128), 256, 0, stream>>>(
      xb, w1bT, b1, nullptr, h1, nullptr, NTOK, DIM, MHID, 0);
  gemm_bf16<<<dim3(DIM / 128, NTOK / 128), 256, 0, stream>>>(
      h1, w2bT, b2, x, nullptr, x2, NTOK, MHID, DIM, 1);

  ln2_k<<<NTOK, 256, 0, stream>>>(x2, ln2g, ln2b, x3, x3b);
  router_k<<<NTOK / 256, 256, 0, stream>>>(x3, wr, gates);

  hipMemsetAsync(outacc, 0, (size_t)NTOK * DIM * 4, stream);
  expert_moe<<<dim3(NTOK / 64, NEXP), 256, 0, stream>>>(
      x3b, ew1bT, eb1, ew2bT, eb2, gates, outacc);

  lnf_k<<<NTOK, 256, 0, stream>>>(outacc, lnfg, lnfb, out);
}
